// GatedDeltaNet_31722628448793
// MI455X (gfx1250) — matrix operand order repaired and checked by reading, not compiled, not run on hardware
//
#include <hip/hip_runtime.h>
#include <hip/hip_bf16.h>

typedef __bf16 bf16_t;
typedef __attribute__((ext_vector_type(16))) __bf16 v16bf;
typedef __attribute__((ext_vector_type(8)))  __bf16 v8bf;
typedef __attribute__((ext_vector_type(8)))  float  v8f;
typedef __attribute__((ext_vector_type(4)))  float  f32x4;
typedef __attribute__((ext_vector_type(2)))  float  v2f;
typedef __attribute__((ext_vector_type(4)))  int    i32x4;
typedef __attribute__((address_space(1))) i32x4* gptr_i32x4;
typedef __attribute__((address_space(3))) i32x4* lptr_i32x4;

#define BB 2
#define TT 4096
#define DD 2048
#define HVn 32
#define HKn 16
#define DKn 128
#define KEYD 2048
#define VALD 4096
#define CONVD 8192
#define NCHUNK 64

#if defined(__has_builtin)
#if __has_builtin(__builtin_amdgcn_global_load_async_to_lds_b128)
#define HAVE_ASYNC_LDS 1
#endif
#endif
#ifndef HAVE_ASYNC_LDS
#define HAVE_ASYNC_LDS 0
#endif

// ---------------------------------------------------------------------------
// f32 WMMA 16x16x4 tile helpers (CDNA5 V_WMMA_F32_16X16X4_F32)
// A fragment: lane&15 = M row, VGPR0/1 = K = (lane<16?0:2)+{0,1}
// B fragment: lane&15 = N col, VGPR0/1 = K = (lane<16?0:2)+{0,1}
// C/D:        lane&15 = N, VGPR r -> M = r + (lane<16?0:8)
// ---------------------------------------------------------------------------
static __device__ inline v8f tile_mm(v8f acc, const float* A, int lda,
                                     const float* Bm, int ldb, int K, int lane) {
  const int m  = lane & 15;
  const int ko = (lane < 16) ? 0 : 2;
  for (int k0 = 0; k0 < K; k0 += 4) {
    v2f a = *(const v2f*)&A[m * lda + k0 + ko];   // 8B-aligned (lda even)
    v2f b;
    b.x = Bm[(k0 + ko) * ldb + m];
    b.y = Bm[(k0 + ko + 1) * ldb + m];
    acc = __builtin_amdgcn_wmma_f32_16x16x4_f32(false, a, false, b, (short)0, acc, false, false);
  }
  return acc;
}

// B given transposed (row-major N x K): C += A * B^T
static __device__ inline v8f tile_mm_bt(v8f acc, const float* A, int lda,
                                        const float* Bt, int ldbt, int K, int lane) {
  const int m  = lane & 15;
  const int ko = (lane < 16) ? 0 : 2;
  for (int k0 = 0; k0 < K; k0 += 4) {
    v2f a = *(const v2f*)&A[m * lda + k0 + ko];
    v2f b = *(const v2f*)&Bt[m * ldbt + k0 + ko];
    acc = __builtin_amdgcn_wmma_f32_16x16x4_f32(false, a, false, b, (short)0, acc, false, false);
  }
  return acc;
}

// A given transposed (stored K x M) and row-scaled: C += (scale[k]*A[k][m])^T * B
static __device__ inline v8f tile_mm_ats(v8f acc, const float* At_, int lda,
                                         const float* scale, const float* Bm, int ldb,
                                         int K, int lane) {
  const int m  = lane & 15;
  const int ko = (lane < 16) ? 0 : 2;
  for (int k0 = 0; k0 < K; k0 += 4) {
    const int k1 = k0 + ko, k2 = k0 + ko + 1;
    v2f a, b;
    a.x = At_[k1 * lda + m] * scale[k1];
    a.y = At_[k2 * lda + m] * scale[k2];
    b.x = Bm[k1 * ldb + m];
    b.y = Bm[k2 * ldb + m];
    acc = __builtin_amdgcn_wmma_f32_16x16x4_f32(false, a, false, b, (short)0, acc, false, false);
  }
  return acc;
}

// ---------------------------------------------------------------------------
// f32 -> bf16 convert
// ---------------------------------------------------------------------------
__global__ void cvt_bf16_kernel(const float* __restrict__ in, bf16_t* __restrict__ out, size_t n) {
  size_t i = (size_t)blockIdx.x * 256 + threadIdx.x;
  if (i < n) out[i] = (bf16_t)in[i];
}

// ---------------------------------------------------------------------------
// bf16 GEMM: C[M,N] = A[M,K] @ B[K,N], f32 accumulate. 64x64 tile / block,
// 8 wave32 waves, each owns one 16-row block and two 16-col blocks.
// A tile staged via GLOBAL_LOAD_ASYNC_TO_LDS_B128 (CDNA5), B staged transposed
// so both WMMA fragments load as two ds_load_b128 each.
// ---------------------------------------------------------------------------
__global__ __launch_bounds__(256) void gemm_bf16_kernel(
    const bf16_t* __restrict__ A, const bf16_t* __restrict__ B,
    float* __restrict__ C, int M, int N, int K) {
  __shared__ bf16_t As[64][40];   // row-major, 80B rows (16B aligned)
  __shared__ bf16_t Bt[64][40];   // transposed: Bt[n][k]
  const int tid  = threadIdx.x;
  const int lane = tid & 31;
  const int wave = tid >> 5;
  const int tm = blockIdx.y * 64;
  const int tn = blockIdx.x * 64;
  const int rb  = wave & 3;
  const int cb0 = (wave >> 2) * 2;

  v8f acc0 = {0,0,0,0,0,0,0,0};
  v8f acc1 = {0,0,0,0,0,0,0,0};

  const int ra = tid >> 2, sa = (tid & 3) * 8;      // A: 16B segment per thread
  const int rB = tid >> 3, cB = (tid & 7) * 8;      // B: 8 elems per thread

  for (int k0 = 0; k0 < K; k0 += 32) {
    // ---- stage A (64x32) ----
    {
      const bf16_t* gp = &A[(size_t)(tm + ra) * K + k0 + sa];
      bf16_t* lp = &As[ra][sa];
#if HAVE_ASYNC_LDS
      __builtin_amdgcn_global_load_async_to_lds_b128(
          (gptr_i32x4)gp, (lptr_i32x4)lp, 0, 0);
#else
      *(f32x4*)lp = *(const f32x4*)gp;
#endif
    }
    // ---- stage B (32x64) transposed into Bt ----
    {
      union { f32x4 v; bf16_t h[8]; } u;
      u.v = *(const f32x4*)&B[(size_t)(k0 + rB) * N + tn + cB];
#pragma unroll
      for (int e = 0; e < 8; ++e) Bt[cB + e][rB] = u.h[e];
    }
    if (k0 + 32 < K) {
      __builtin_prefetch(&A[(size_t)(tm + ra) * K + k0 + 32 + sa], 0, 1);
      __builtin_prefetch(&B[(size_t)(k0 + 32 + rB) * N + tn + cB], 0, 1);
    }
#if HAVE_ASYNC_LDS
#if __has_builtin(__builtin_amdgcn_s_wait_asynccnt)
    __builtin_amdgcn_s_wait_asynccnt(0);
#else
    asm volatile("s_wait_asynccnt 0x0" ::: "memory");
#endif
#endif
    __syncthreads();

    // ---- fragments: two 16B LDS loads each ----
    const int m = lane & 15;
    const int kbase = (lane < 16) ? 0 : 8;    // A: K pairs per ISA layout
    v8bf alo = *(const v8bf*)&As[rb * 16 + m][kbase];
    v8bf ahi = *(const v8bf*)&As[rb * 16 + m][kbase + 16];
    v16bf af = __builtin_shufflevector(alo, ahi, 0,1,2,3,4,5,6,7,8,9,10,11,12,13,14,15);

    const int koff = (lane < 16) ? 0 : 8;     // B: K pairs per ISA layout (same as A)
    const int n0 = cb0 * 16 + m;
    v8bf b0lo = *(const v8bf*)&Bt[n0][koff];
    v8bf b0hi = *(const v8bf*)&Bt[n0][koff + 16];
    v16bf bf0 = __builtin_shufflevector(b0lo, b0hi, 0,1,2,3,4,5,6,7,8,9,10,11,12,13,14,15);
    v8bf b1lo = *(const v8bf*)&Bt[n0 + 16][koff];
    v8bf b1hi = *(const v8bf*)&Bt[n0 + 16][koff + 16];
    v16bf bf1 = __builtin_shufflevector(b1lo, b1hi, 0,1,2,3,4,5,6,7,8,9,10,11,12,13,14,15);

    acc0 = __builtin_amdgcn_wmma_f32_16x16x32_bf16(false, af, false, bf0, (short)0, acc0, false, false);
    acc1 = __builtin_amdgcn_wmma_f32_16x16x32_bf16(false, af, false, bf1, (short)0, acc1, false, false);
    __syncthreads();
  }

#pragma unroll
  for (int r = 0; r < 8; ++r) {
    int mm = tm + rb * 16 + r + ((lane < 16) ? 0 : 8);
    int nn = tn + cb0 * 16 + (lane & 15);
    C[(size_t)mm * N + nn]      = acc0[r];
    C[(size_t)mm * N + nn + 16] = acc1[r];
  }
}

// ---------------------------------------------------------------------------
// beta / g gate kernel: one block per token, dots over D=2048 for 32 heads.
// ---------------------------------------------------------------------------
__global__ __launch_bounds__(64) void ba_kernel(
    const float* __restrict__ x, const float* __restrict__ Wb, const float* __restrict__ Wa,
    const float* __restrict__ Alog, const float* __restrict__ dtb,
    float* __restrict__ beta, float* __restrict__ g) {
  __shared__ float xr[DD];
  const size_t tok = blockIdx.x;
  for (int i = threadIdx.x * 4; i < DD; i += 64 * 4)
    *(f32x4*)&xr[i] = *(const f32x4*)&x[tok * DD + i];
  __syncthreads();
  const int tid = threadIdx.x;
  if (tid < HVn) {
    float bb = 0.f, aa = 0.f;
    for (int d = 0; d < DD; ++d) {
      float xv = xr[d];
      bb += xv * Wb[d * HVn + tid];
      aa += xv * Wa[d * HVn + tid];
    }
    beta[tok * HVn + tid] = 1.f / (1.f + __expf(-bb));
    float si = aa + dtb[tid];
    float sp = (si > 20.f) ? si : log1pf(__expf(si));
    g[tok * HVn + tid] = -__expf(Alog[tid]) * sp;
  }
}

// ---------------------------------------------------------------------------
// causal depthwise conv(K=4) + SiLU + split + per-head L2 norm (q,k) + q scale
// ---------------------------------------------------------------------------
__global__ __launch_bounds__(256) void conv_kernel(
    const float* __restrict__ mixed, const float* __restrict__ convw,
    float* __restrict__ qn, float* __restrict__ kn, float* __restrict__ vv) {
  __shared__ float buf[CONVD];
  __shared__ float red[256];
  __shared__ float normf[32];
  const size_t tok = blockIdx.x;
  const int b = (int)(tok / TT);
  const int t = (int)(tok % TT);
  const int tid = threadIdx.x;

  for (int j = 0; j < CONVD / 256; ++j) {
    int c = tid + 256 * j;
    float acc = 0.f;
#pragma unroll
    for (int kk = 0; kk < 4; ++kk) {
      int ts = t - 3 + kk;
      if (ts >= 0) acc += convw[c * 4 + kk] * mixed[((size_t)b * TT + ts) * CONVD + c];
    }
    buf[c] = acc / (1.f + __expf(-acc));   // SiLU
  }
  __syncthreads();
  {
    int head = tid >> 3, part = tid & 7;
    float s = 0.f;
    int base = head * 128 + part * 16;
#pragma unroll
    for (int e = 0; e < 16; ++e) { float v = buf[base + e]; s += v * v; }
    red[tid] = s;
  }
  __syncthreads();
  if (tid < 32) {
    float s = 0.f;
#pragma unroll
    for (int p = 0; p < 8; ++p) s += red[tid * 8 + p];
    normf[tid] = rsqrtf(s + 1e-6f);
  }
  __syncthreads();
  const float qscale = 0.08838834764831845f;  // DK^-0.5
  for (int j = 0; j < CONVD / 256; ++j) {
    int c = tid + 256 * j;
    float v = buf[c];
    if (c < KEYD) {
      qn[tok * KEYD + c] = v * normf[c >> 7] * qscale;
    } else if (c < 2 * KEYD) {
      int cc = c - KEYD;
      kn[tok * KEYD + cc] = v * normf[16 + (cc >> 7)];
    } else {
      int cc = c - 2 * KEYD;
      vv[tok * VALD + cc] = v;
    }
  }
}

// ---------------------------------------------------------------------------
// per-chunk cumsum of g
// ---------------------------------------------------------------------------
__global__ __launch_bounds__(64) void cumsum_kernel(const float* __restrict__ g,
                                                    float* __restrict__ gcum) {
  const int blk = blockIdx.x;             // ((b*32+h)*NCHUNK + c)
  const int c = blk % NCHUNK;
  const int bh = blk / NCHUNK;
  const int h = bh & 31, b = bh >> 5;
  const int t0 = c * 64;
  float s = 0.f;
  for (int j = 0; j <= (int)threadIdx.x; ++j)
    s += g[((size_t)(b * TT) + t0 + j) * HVn + h];
  gcum[((size_t)(b * TT) + t0 + threadIdx.x) * HVn + h] = s;
}

// ---------------------------------------------------------------------------
// chunked gated delta rule core. One block per (b, head): 256 threads, 8 waves.
// All chunk tiles + 128x128 f32 state live in LDS (~215 KB, CDNA5 WGP only).
// All matmuls via V_WMMA_F32_16X16X4_F32.
// ---------------------------------------------------------------------------
#define SP 132   // padded stride for 128-wide LDS tiles (even -> 8B aligned pairs)
#define AP 68    // padded stride for 64-wide attn matrix
#define CORE_LDS_FLOATS (128*SP + 4*64*SP + 64*AP + 64 + 64)
#define CORE_LDS_BYTES  (CORE_LDS_FLOATS * 4)

__global__ __launch_bounds__(256) void core_kernel(
    const float* __restrict__ qn, const float* __restrict__ kn,
    const float* __restrict__ vv, const float* __restrict__ beta,
    const float* __restrict__ gcum, float* __restrict__ core) {
  extern __shared__ float sm[];
  float* S  = sm;                 // 128 x SP  running state (DK x DV)
  float* Qs = S  + 128 * SP;      // 64 x SP
  float* Ks = Qs + 64 * SP;       // 64 x SP
  float* KB = Ks + 64 * SP;       // 64 x SP   k_beta -> k_cumdecay
  float* Vs = KB + 64 * SP;       // 64 x SP   v_beta -> v_corrected -> v_new
  float* At = Vs + 64 * SP;       // 64 x AP   attn / intra
  float* Gc = At + 64 * AP;       // 64        cumulative g
  float* Dd = Gc + 64;            // 64        beta, then decay factors

  const int tid  = threadIdx.x;
  const int lane = tid & 31;
  const int wave = tid >> 5;
  const int bh = blockIdx.x;
  const int b = bh >> 5, h = bh & 31, hk = h >> 1;   // GQA: 2 v-heads per kv-head

  for (int i = tid; i < 128 * SP; i += 256) S[i] = 0.f;
  __syncthreads();

  for (int c = 0; c < NCHUNK; ++c) {
    const int t0 = c * 64;
    if (tid < 64) {
      size_t idx = ((size_t)(b * TT + t0 + tid)) * HVn + h;
      Gc[tid] = gcum[idx];
      Dd[tid] = beta[idx];
    }
    __syncthreads();
    // vectorized staging: 64 rows x 128 cols as float4
    for (int i = tid; i < 64 * 32; i += 256) {
      int r = i >> 5, d = (i & 31) * 4;
      size_t tq = ((size_t)(b * TT + t0 + r) * HKn + hk) * DKn + d;
      size_t tv = ((size_t)(b * TT + t0 + r) * HVn + h) * DKn + d;
      f32x4 kv = *(const f32x4*)&kn[tq];
      float bs = Dd[r];
      *(f32x4*)&Qs[r * SP + d] = *(const f32x4*)&qn[tq];
      *(f32x4*)&Ks[r * SP + d] = kv;
      *(f32x4*)&KB[r * SP + d] = kv * bs;
      *(f32x4*)&Vs[r * SP + d] = (*(const f32x4*)&vv[tv]) * bs;
    }
    __syncthreads();

    // ---- attn = -(k_beta @ k^T) * decay, strictly lower triangular ----
    for (int tp = 0; tp < 2; ++tp) {
      int tl = wave * 2 + tp;
      int I = tl >> 2, J = tl & 3;
      if (J > I) {
#pragma unroll
        for (int r = 0; r < 8; ++r) {
          int mm = I * 16 + r + ((lane < 16) ? 0 : 8);
          int nn = J * 16 + (lane & 15);
          At[mm * AP + nn] = 0.f;
        }
      } else {
        v8f acc = {0,0,0,0,0,0,0,0};
        acc = tile_mm_bt(acc, &KB[(I * 16) * SP], SP, &Ks[(J * 16) * SP], SP, 128, lane);
#pragma unroll
        for (int r = 0; r < 8; ++r) {
          int mm = I * 16 + r + ((lane < 16) ? 0 : 8);
          int nn = J * 16 + (lane & 15);
          At[mm * AP + nn] = (mm > nn) ? (-acc[r] * __expf(Gc[mm] - Gc[nn])) : 0.f;
        }
      }
    }
    __syncthreads();

    // ---- forward substitution: row_i += row_i @ A (sequential) ----
    for (int i = 1; i < 64; ++i) {
      float nv = 0.f;
      if (tid < i) {
        nv = At[i * AP + tid];
        for (int l = tid + 1; l < i; ++l)
          nv += At[i * AP + l] * At[l * AP + tid];
      }
      __syncthreads();
      if (tid < i) At[i * AP + tid] = nv;
      __syncthreads();
    }
    if (tid < 64) At[tid * AP + tid] = 1.f;   // + I
    __syncthreads();

    // ---- v_corrected = attn @ v_beta (in place, descending row blocks) ----
    for (int I = 3; I >= 0; --I) {
      v8f acc = {0,0,0,0,0,0,0,0};
      for (int J = 0; J <= I; ++J)
        acc = tile_mm(acc, &At[(I * 16) * AP + J * 16], AP, &Vs[(J * 16) * SP + wave * 16], SP, 16, lane);
      __syncthreads();
#pragma unroll
      for (int r = 0; r < 8; ++r) {
        int mm = I * 16 + r + ((lane < 16) ? 0 : 8);
        Vs[mm * SP + wave * 16 + (lane & 15)] = acc[r];
      }
      __syncthreads();
    }

    // ---- k_cumdecay = attn @ (k_beta * exp(g)) (in place in KB) ----
    for (int i = tid; i < 64 * 128; i += 256) {
      int r = i >> 7, d = i & 127;
      KB[r * SP + d] *= __expf(Gc[r]);
    }
    __syncthreads();
    for (int I = 3; I >= 0; --I) {
      v8f acc = {0,0,0,0,0,0,0,0};
      for (int J = 0; J <= I; ++J)
        acc = tile_mm(acc, &At[(I * 16) * AP + J * 16], AP, &KB[(J * 16) * SP + wave * 16], SP, 16, lane);
      __syncthreads();
#pragma unroll
      for (int r = 0; r < 8; ++r) {
        int mm = I * 16 + r + ((lane < 16) ? 0 : 8);
        KB[mm * SP + wave * 16 + (lane & 15)] = acc[r];
      }
      __syncthreads();
    }

    // ---- v_new = v_corrected - k_cumdecay @ S ----
    for (int I = 0; I < 4; ++I) {
      v8f acc = {0,0,0,0,0,0,0,0};
      acc = tile_mm(acc, &KB[(I * 16) * SP], SP, &S[wave * 16], SP, 128, lane);
#pragma unroll
      for (int r = 0; r < 8; ++r) {
        int mm = I * 16 + r + ((lane < 16) ? 0 : 8);
        Vs[mm * SP + wave * 16 + (lane & 15)] -= acc[r];
      }
    }
    __syncthreads();

    // ---- intra = (q @ k^T) * decay, lower incl diag (overwrite At) ----
    for (int tp = 0; tp < 2; ++tp) {
      int tl = wave * 2 + tp;
      int I = tl >> 2, J = tl & 3;
      if (J > I) {
#pragma unroll
        for (int r = 0; r < 8; ++r) {
          int mm = I * 16 + r + ((lane < 16) ? 0 : 8);
          int nn = J * 16 + (lane & 15);
          At[mm * AP + nn] = 0.f;
        }
      } else {
        v8f acc = {0,0,0,0,0,0,0,0};
        acc = tile_mm_bt(acc, &Qs[(I * 16) * SP], SP, &Ks[(J * 16) * SP], SP, 128, lane);
#pragma unroll
        for (int r = 0; r < 8; ++r) {
          int mm = I * 16 + r + ((lane < 16) ? 0 : 8);
          int nn = J * 16 + (lane & 15);
          At[mm * AP + nn] = (mm >= nn) ? (acc[r] * __expf(Gc[mm] - Gc[nn])) : 0.f;
        }
      }
    }
    __syncthreads();

    // ---- out = exp(g) * (q @ S) + intra @ v_new ----
    for (int I = 0; I < 4; ++I) {
      v8f acc = {0,0,0,0,0,0,0,0};
      acc = tile_mm(acc, &Qs[(I * 16) * SP], SP, &S[wave * 16], SP, 128, lane);
      v8f acc2 = {0,0,0,0,0,0,0,0};
      for (int J = 0; J <= I; ++J)
        acc2 = tile_mm(acc2, &At[(I * 16) * AP + J * 16], AP, &Vs[(J * 16) * SP + wave * 16], SP, 16, lane);
#pragma unroll
      for (int r = 0; r < 8; ++r) {
        int mm = I * 16 + r + ((lane < 16) ? 0 : 8);
        int nn = wave * 16 + (lane & 15);
        core[((size_t)(b * TT + t0 + mm) * HVn + h) * DKn + nn] =
            __expf(Gc[mm]) * acc[r] + acc2[r];
      }
    }

    // ---- state: S = S*exp(g_last) + (k * exp(g_last - g))^T @ v_new ----
    if (tid < 64) Dd[tid] = __expf(Gc[63] - Gc[tid]);
    __syncthreads();
    const float egl = __expf(Gc[63]);
    for (int vb = 0; vb < 8; ++vb) {
      v8f acc = {0,0,0,0,0,0,0,0};
      acc = tile_mm_ats(acc, &Ks[wave * 16], SP, Dd, &Vs[vb * 16], SP, 64, lane);
#pragma unroll
      for (int r = 0; r < 8; ++r) {
        int dd = wave * 16 + r + ((lane < 16) ? 0 : 8);
        int nn = vb * 16 + (lane & 15);
        size_t si = (size_t)dd * SP + nn;
        S[si] = S[si] * egl + acc[r];
      }
    }
    __syncthreads();
  }
}

// ---------------------------------------------------------------------------
// gated RMSNorm: x = core * silu(z); x * rsqrt(mean(x^2)+eps) * w  -> bf16
// ---------------------------------------------------------------------------
__global__ __launch_bounds__(128) void rmsnorm_kernel(
    const float* __restrict__ corein, const float* __restrict__ z,
    const float* __restrict__ w, bf16_t* __restrict__ outb) {
  __shared__ float red[128];
  const size_t row = blockIdx.x;
  const int tid = threadIdx.x;
  float zz = z[row * 128 + tid];
  float xx = corein[row * 128 + tid] * (zz / (1.f + __expf(-zz)));
  red[tid] = xx * xx;
  __syncthreads();
  for (int s = 64; s > 0; s >>= 1) {
    if (tid < s) red[tid] += red[tid + s];
    __syncthreads();
  }
  float inv = rsqrtf(red[0] * (1.f / 128.f) + 1e-6f);
  outb[row * 128 + tid] = (bf16_t)(xx * inv * w[tid]);
}

// ---------------------------------------------------------------------------
extern "C" void kernel_launch(void* const* d_in, const int* in_sizes, int n_in,
                              void* d_out, int out_size, void* d_ws, size_t ws_size,
                              hipStream_t stream) {
  (void)in_sizes; (void)n_in; (void)out_size; (void)ws_size;
  const float* x     = (const float*)d_in[0];
  const float* Wqkv  = (const float*)d_in[1];
  const float* Wz    = (const float*)d_in[2];
  const float* Wb    = (const float*)d_in[3];
  const float* Wa    = (const float*)d_in[4];
  const float* convw = (const float*)d_in[5];
  const float* dtb   = (const float*)d_in[6];
  const float* Alog  = (const float*)d_in[7];
  const float* normw = (const float*)d_in[8];
  const float* Wout  = (const float*)d_in[9];
  float* out = (float*)d_out;

  const size_t MT = (size_t)BB * TT;   // 8192 tokens
  char* ws = (char*)d_ws;
  size_t off = 0;
  auto alloc = [&](size_t bytes) { size_t o = off; off += (bytes + 255) & ~(size_t)255; return o; };

  bf16_t* xb    = (bf16_t*)(ws + alloc(MT * DD * 2));
  bf16_t* wqkvb = (bf16_t*)(ws + alloc((size_t)DD * CONVD * 2));
  bf16_t* wzb   = (bf16_t*)(ws + alloc((size_t)DD * VALD * 2));
  bf16_t* woutb = (bf16_t*)(ws + alloc((size_t)VALD * DD * 2));
  float*  mixed = (float*)(ws + alloc(MT * CONVD * 4));
  float*  zbuf  = (float*)(ws + alloc(MT * VALD * 4));
  float*  qnb   = (float*)(ws + alloc(MT * KEYD * 4));
  float*  knb   = (float*)(ws + alloc(MT * KEYD * 4));
  float*  vvb   = (float*)(ws + alloc(MT * VALD * 4));
  float*  corb  = (float*)(ws + alloc(MT * VALD * 4));
  float*  betab = (float*)(ws + alloc(MT * HVn * 4));
  float*  gb    = (float*)(ws + alloc(MT * HVn * 4));
  float*  gcb   = (float*)(ws + alloc(MT * HVn * 4));
  bf16_t* nrmb  = (bf16_t*)(ws + alloc(MT * VALD * 2));

  (void)hipFuncSetAttribute((const void*)core_kernel,
                            hipFuncAttributeMaxDynamicSharedMemorySize, CORE_LDS_BYTES);

  // bf16 conversions
  {
    size_t n;
    n = MT * DD;              cvt_bf16_kernel<<<dim3((n + 255) / 256), 256, 0, stream>>>(x,    xb,    n);
    n = (size_t)DD * CONVD;   cvt_bf16_kernel<<<dim3((n + 255) / 256), 256, 0, stream>>>(Wqkv, wqkvb, n);
    n = (size_t)DD * VALD;    cvt_bf16_kernel<<<dim3((n + 255) / 256), 256, 0, stream>>>(Wz,   wzb,   n);
    n = (size_t)VALD * DD;    cvt_bf16_kernel<<<dim3((n + 255) / 256), 256, 0, stream>>>(Wout, woutb, n);
  }

  // mixed = x @ W_qkv ; z = x @ W_z  (bf16 WMMA)
  gemm_bf16_kernel<<<dim3(CONVD / 64, MT / 64), 256, 0, stream>>>(xb, wqkvb, mixed, (int)MT, CONVD, DD);
  gemm_bf16_kernel<<<dim3(VALD / 64, MT / 64), 256, 0, stream>>>(xb, wzb, zbuf, (int)MT, VALD, DD);

  // beta / g gates (f32)
  ba_kernel<<<dim3((unsigned)MT), 64, 0, stream>>>(x, Wb, Wa, Alog, dtb, betab, gb);

  // conv + silu + split + l2norm
  conv_kernel<<<dim3((unsigned)MT), 256, 0, stream>>>(mixed, convw, qnb, knb, vvb);

  // per-chunk cumsum of g
  cumsum_kernel<<<dim3(BB * HVn * NCHUNK), 64, 0, stream>>>(gb, gcb);

  // delta-rule core (f32 WMMA, state in 320KB WGP LDS)
  core_kernel<<<dim3(BB * HVn), 256, CORE_LDS_BYTES, stream>>>(qnb, knb, vvb, betab, gcb, corb);

  // gated rmsnorm -> bf16
  rmsnorm_kernel<<<dim3((unsigned)(MT * HVn)), 128, 0, stream>>>(corb, zbuf, normw, nrmb);

  // out = normed @ W_out (bf16 WMMA, f32 out)
  gemm_bf16_kernel<<<dim3(DD / 64, MT / 64), 256, 0, stream>>>(nrmb, woutb, out, (int)MT, DD, VALD);
}
